// DocumentRQVAE_17171279249691
// MI455X (gfx1250) — compile-verified, matrix operand order checked by reading, not run on hardware
//
#include <hip/hip_runtime.h>
#include <stdint.h>

// ---------------------------------------------------------------------------
// DocumentRQVAE forward for MI455X (gfx1250, wave32).
//  - All large matmuls: v_wmma_f32_16x16x32_bf16, f32 accumulate.
//  - Weights converted once to a bf16 arena; GEMM B-tiles are moved with
//    GLOBAL_LOAD_ASYNC_TO_LDS_B128 (ASYNCcnt) -- no VGPR round trip.
//  - GEMM is software-pipelined with double-buffered LDS (BK=64).
//  - Attention is flash-style; online softmax done in the WMMA C layout.
//  - Cross-attention over the length-1 VQ memory collapses to a broadcast.
// ---------------------------------------------------------------------------

typedef __bf16 bf16_t;
typedef bf16_t v16bf __attribute__((ext_vector_type(16)));
typedef float  v8f   __attribute__((ext_vector_type(8)));

#define NTOK   16384      // B*T
#define Bb     32
#define Tt     512
#define DIm    768
#define Dm     512
#define DFm    2048
#define Hh     8
#define Dh     64
#define Kcb    1024
#define NCb    4
#define HIDm   128
#define COMMITF 0.25f

__device__ inline v8f v8zero() {
  v8f v;
#pragma unroll
  for (int i = 0; i < 8; ++i) v[i] = 0.0f;
  return v;
}

// 16x32 bf16 A/B fragment from an LDS row pointer (offset to K-base).
// ISA 7.12.2: lanes 0-15 take K chunks {0..7,16..23}, lanes 16-31 {8..15,24..31}.
__device__ inline v16bf ld_frag(const bf16_t* rowp, int lane) {
  const int hf = (lane >> 4) * 8;
  v16bf r;
#pragma unroll
  for (int i = 0; i < 8; ++i) {
    r[i]     = rowp[hf + i];
    r[i + 8] = rowp[hf + 16 + i];
  }
  return r;
}

__device__ inline v8f wmma_bf16(v16bf a, v16bf b, v8f c) {
  return __builtin_amdgcn_wmma_f32_16x16x32_bf16(
      false, a, false, b, (short)0, c, false, false);
}

__device__ inline float wave_sum(float v) {
#pragma unroll
  for (int m = 16; m > 0; m >>= 1) v += __shfl_xor(v, m, 32);
  return v;
}

// generic-pointer -> LDS byte offset (ISA 10.2: LDS aperture addr = addr[31:0])
__device__ inline unsigned lds_off32(const void* p) {
  return (unsigned)(uintptr_t)p;
}

__device__ inline void async_b128_to_lds(unsigned lds, const bf16_t* gp) {
  unsigned long long ga = (unsigned long long)(uintptr_t)gp;
  asm volatile("global_load_async_to_lds_b128 %0, %1, off"
               :: "v"(lds), "v"(ga) : "memory");
}
__device__ inline void wait_async0() {
  asm volatile("s_wait_asynccnt 0x0" ::: "memory");
}

// ---------------------------------------------------------------------------
// GEMM:  C[M,N] = act(A[M,K] @ W[N,K]^T + bias), W pre-converted to bf16.
// BM=128, BN=64, BK=64, 256 threads (8 waves as 4x2), wave tile 32x32.
// Double-buffered LDS; A staged through registers (batched loads), B moved
// with async-to-LDS. Requires M%128==0, N%64==0, K%64==0 (all call sites).
// act: 0=none, 1=relu, 2=tanh
// ---------------------------------------------------------------------------
#define GBM 128
#define GBN 64
#define GBK 64
#define GLP 72   // padded LDS row (bf16) -> conflict-free fragment reads

__global__ __launch_bounds__(256) void gemm_nt(
    const float* __restrict__ A, const bf16_t* __restrict__ Wb,
    const float* __restrict__ bias, float* __restrict__ C,
    int M, int N, int K, int act) {
  __shared__ bf16_t sA[2][GBM][GLP];   // 36 KB
  __shared__ bf16_t sB[2][GBN][GLP];   // 18 KB

  const int tid  = threadIdx.x;
  const int wave = tid >> 5, lane = tid & 31;
  const int wm = wave >> 1, wn = wave & 1;
  const int bm = blockIdx.x * GBM;
  const int bn = blockIdx.y * GBN;
  const int m15 = lane & 15;
  const int lhalf = lane >> 4;

  v8f acc[2][2];
#pragma unroll
  for (int i = 0; i < 2; ++i)
#pragma unroll
    for (int j = 0; j < 2; ++j) acc[i][j] = v8zero();

  float4 areg[8];  // 128x64 f32 tile = 2048 float4 / 256 thr = 8 each

  auto issueA = [&](int kb) {
#pragma unroll
    for (int i = 0; i < 8; ++i) {
      const int idx = tid + i * 256;
      const int r = idx >> 4, c = (idx & 15) * 4;
      areg[i] = *(const float4*)(A + (size_t)(bm + r) * K + kb + c);
    }
  };
  auto issueB = [&](int kb, int buf) {  // 64x64 bf16 = 512 x 16B chunks, 2 each
#pragma unroll
    for (int i = 0; i < 2; ++i) {
      const int idx = tid + i * 256;
      const int r = idx >> 3, c8 = (idx & 7) * 8;
      async_b128_to_lds(lds_off32(&sB[buf][r][c8]),
                        Wb + (size_t)(bn + r) * K + kb + c8);
    }
  };
  auto storeA = [&](int buf) {
#pragma unroll
    for (int i = 0; i < 8; ++i) {
      const int idx = tid + i * 256;
      const int r = idx >> 4, c = (idx & 15) * 4;
      sA[buf][r][c + 0] = (bf16_t)areg[i].x;
      sA[buf][r][c + 1] = (bf16_t)areg[i].y;
      sA[buf][r][c + 2] = (bf16_t)areg[i].z;
      sA[buf][r][c + 3] = (bf16_t)areg[i].w;
    }
  };

  // prologue: fill buffer 0
  issueA(0);
  issueB(0, 0);
  storeA(0);
  wait_async0();
  __syncthreads();

  int cur = 0;
  for (int kb = 0; kb < K; kb += GBK) {
    const int nkb = kb + GBK;
    const bool has_next = nkb < K;
    if (has_next) {
      issueA(nkb);          // global loads in flight during compute
      issueB(nkb, cur ^ 1); // async engine fills next B buffer
    }
#pragma unroll
    for (int ks = 0; ks < GBK; ks += 32) {
      v16bf af0 = ld_frag(&sA[cur][wm * 32 + m15][ks], lane);
      v16bf af1 = ld_frag(&sA[cur][wm * 32 + 16 + m15][ks], lane);
      v16bf bf0 = ld_frag(&sB[cur][wn * 32 + m15][ks], lane);
      v16bf bf1 = ld_frag(&sB[cur][wn * 32 + 16 + m15][ks], lane);
      acc[0][0] = wmma_bf16(af0, bf0, acc[0][0]);
      acc[0][1] = wmma_bf16(af0, bf1, acc[0][1]);
      acc[1][0] = wmma_bf16(af1, bf0, acc[1][0]);
      acc[1][1] = wmma_bf16(af1, bf1, acc[1][1]);
    }
    if (has_next) {
      storeA(cur ^ 1);      // waits for areg loads here, after compute
      wait_async0();
    }
    __syncthreads();
    cur ^= 1;
  }

  // epilogue: C fragment layout -> lane l, VGPR v: row = v + (l>=16)*8
#pragma unroll
  for (int i = 0; i < 2; ++i) {
#pragma unroll
    for (int j = 0; j < 2; ++j) {
      const int col = bn + wn * 32 + j * 16 + m15;
      const float bv = bias ? bias[col] : 0.0f;
      const int row0 = bm + wm * 32 + i * 16 + lhalf * 8;
#pragma unroll
      for (int v = 0; v < 8; ++v) {
        float val = acc[i][j][v] + bv;
        if (act == 1) val = fmaxf(val, 0.0f);
        else if (act == 2) val = tanhf(val);
        C[(size_t)(row0 + v) * N + col] = val;
      }
    }
  }
}

// ---------------------------------------------------------------------------
// Flash attention, one (b,h,query-tile-of-128) per block, 256 threads.
// ---------------------------------------------------------------------------
#define FQT 128
#define FKT 64
#define FLP 72

__global__ __launch_bounds__(256) void flash_attn(
    const float* __restrict__ Q, const float* __restrict__ K,
    const float* __restrict__ V, float* __restrict__ O) {
  __shared__ bf16_t sQ[FQT][FLP];
  __shared__ bf16_t sK[FKT][FLP];
  __shared__ bf16_t sVt[Dh][FLP];   // V transposed: [dim][key]
  __shared__ bf16_t sP[FQT][FLP];

  const int qt = blockIdx.x, h = blockIdx.y, b = blockIdx.z;
  const size_t ld = Hh * Dh;  // 512
  const size_t base = (size_t)b * Tt * ld + (size_t)h * Dh;
  const float* Qp = Q + base + (size_t)qt * FQT * ld;
  const float* Kp = K + base;
  const float* Vp = V + base;
  float*       Op = O + base + (size_t)qt * FQT * ld;

  const int tid = threadIdx.x;
  const int wave = tid >> 5, lane = tid & 31;
  const int m15 = lane & 15, lhalf = lane >> 4;
  const float scale = 0.125f;

  // stage Q (batched: 8 loads in flight, then convert/store)
  {
    float4 qreg[8];
#pragma unroll
    for (int i = 0; i < 8; ++i) {
      const int idx = tid + i * 256;
      const int r = idx >> 4, c = (idx & 15) * 4;
      qreg[i] = *(const float4*)(Qp + (size_t)r * ld + c);
    }
#pragma unroll
    for (int i = 0; i < 8; ++i) {
      const int idx = tid + i * 256;
      const int r = idx >> 4, c = (idx & 15) * 4;
      sQ[r][c + 0] = (bf16_t)qreg[i].x; sQ[r][c + 1] = (bf16_t)qreg[i].y;
      sQ[r][c + 2] = (bf16_t)qreg[i].z; sQ[r][c + 3] = (bf16_t)qreg[i].w;
    }
  }

  float Mrun[8], Lrun[8];
  v8f Oacc[4];
#pragma unroll
  for (int r = 0; r < 8; ++r) { Mrun[r] = -3.0e38f; Lrun[r] = 0.0f; }
#pragma unroll
  for (int j = 0; j < 4; ++j) Oacc[j] = v8zero();

  for (int kt = 0; kt < Tt; kt += FKT) {
    // batched staging: 8 b128 loads in flight (4 K-tile + 4 V-tile)
    float4 kreg[4], vreg[4];
#pragma unroll
    for (int i = 0; i < 4; ++i) {
      const int idx = tid + i * 256;
      const int r = idx >> 4, c = (idx & 15) * 4;
      kreg[i] = *(const float4*)(Kp + (size_t)(kt + r) * ld + c);
      vreg[i] = *(const float4*)(Vp + (size_t)(kt + r) * ld + c);
    }
#pragma unroll
    for (int i = 0; i < 4; ++i) {
      const int idx = tid + i * 256;
      const int r = idx >> 4, c = (idx & 15) * 4;
      sK[r][c + 0] = (bf16_t)kreg[i].x; sK[r][c + 1] = (bf16_t)kreg[i].y;
      sK[r][c + 2] = (bf16_t)kreg[i].z; sK[r][c + 3] = (bf16_t)kreg[i].w;
      sVt[c + 0][r] = (bf16_t)vreg[i].x; sVt[c + 1][r] = (bf16_t)vreg[i].y;
      sVt[c + 2][r] = (bf16_t)vreg[i].z; sVt[c + 3][r] = (bf16_t)vreg[i].w;
    }
    __syncthreads();

    // S = Q @ K^T (wave's 16 rows x 64 keys)
    v8f S[4];
#pragma unroll
    for (int j = 0; j < 4; ++j) S[j] = v8zero();
#pragma unroll
    for (int ks = 0; ks < Dh; ks += 32) {
      v16bf aq = ld_frag(&sQ[wave * 16 + m15][ks], lane);
#pragma unroll
      for (int j = 0; j < 4; ++j) {
        v16bf bk = ld_frag(&sK[j * 16 + m15][ks], lane);
        S[j] = wmma_bf16(aq, bk, S[j]);
      }
    }

    // online softmax per row (row = r + lhalf*8 of wave's tile)
#pragma unroll
    for (int r = 0; r < 8; ++r) {
      float rowmax = -3.0e38f;
#pragma unroll
      for (int j = 0; j < 4; ++j) rowmax = fmaxf(rowmax, S[j][r]);
#pragma unroll
      for (int m = 1; m < 16; m <<= 1)
        rowmax = fmaxf(rowmax, __shfl_xor(rowmax, m, 32));
      rowmax *= scale;
      const float mnew = fmaxf(Mrun[r], rowmax);
      const float alpha = __expf(Mrun[r] - mnew);
      float lsum = 0.0f;
#pragma unroll
      for (int j = 0; j < 4; ++j) {
        const float p = __expf(S[j][r] * scale - mnew);
        S[j][r] = p;
        lsum += p;
      }
#pragma unroll
      for (int m = 1; m < 16; m <<= 1) lsum += __shfl_xor(lsum, m, 32);
      Lrun[r] = Lrun[r] * alpha + lsum;
      Mrun[r] = mnew;
#pragma unroll
      for (int j = 0; j < 4; ++j) Oacc[j][r] *= alpha;
    }

    // spill P to wave-private LDS rows to re-enter A-fragment layout
#pragma unroll
    for (int j = 0; j < 4; ++j)
#pragma unroll
      for (int r = 0; r < 8; ++r)
        sP[wave * 16 + r + lhalf * 8][j * 16 + m15] = (bf16_t)S[j][r];

    // O += P @ V
#pragma unroll
    for (int ks = 0; ks < FKT; ks += 32) {
      v16bf ap = ld_frag(&sP[wave * 16 + m15][ks], lane);
#pragma unroll
      for (int j = 0; j < 4; ++j) {
        v16bf bv = ld_frag(&sVt[j * 16 + m15][ks], lane);
        Oacc[j] = wmma_bf16(ap, bv, Oacc[j]);
      }
    }
    __syncthreads();
  }

#pragma unroll
  for (int j = 0; j < 4; ++j)
#pragma unroll
    for (int r = 0; r < 8; ++r) {
      const int row = wave * 16 + r + lhalf * 8;
      Op[(size_t)row * ld + j * 16 + m15] = Oacc[j][r] / Lrun[r];
    }
}

// ---------------------------------------------------------------------------
// LayerNorm over D=512, one block per row, optional fused residual.
// ---------------------------------------------------------------------------
__global__ __launch_bounds__(256) void ln512(
    const float* __restrict__ x, const float* __restrict__ res,
    const float* __restrict__ g, const float* __restrict__ bta,
    float* __restrict__ out) {
  const size_t row = blockIdx.x;
  const int tid = threadIdx.x;
  float v0 = x[row * Dm + tid], v1 = x[row * Dm + 256 + tid];
  if (res) { v0 += res[row * Dm + tid]; v1 += res[row * Dm + 256 + tid]; }
  float s = v0 + v1, sq = v0 * v0 + v1 * v1;
  s = wave_sum(s); sq = wave_sum(sq);
  __shared__ float ss[8], ssq[8];
  __shared__ float mu_s, rs_s;
  if ((tid & 31) == 0) { ss[tid >> 5] = s; ssq[tid >> 5] = sq; }
  __syncthreads();
  if (tid == 0) {
    float S = 0, Q = 0;
#pragma unroll
    for (int i = 0; i < 8; ++i) { S += ss[i]; Q += ssq[i]; }
    const float mu = S / (float)Dm;
    mu_s = mu;
    rs_s = rsqrtf(Q / (float)Dm - mu * mu + 1e-5f);
  }
  __syncthreads();
  const float mu = mu_s, rs = rs_s;
  out[row * Dm + tid]       = (v0 - mu) * rs * g[tid] + bta[tid];
  out[row * Dm + 256 + tid] = (v1 - mu) * rs * g[tid + 256] + bta[tid + 256];
}

// s[row] = dot(x[row, :128], w) + b  (one wave per row)
__global__ __launch_bounds__(256) void row_dot128(
    const float* __restrict__ x, const float* __restrict__ w,
    const float* __restrict__ b, float* __restrict__ out) {
  const int row = blockIdx.x * 8 + (threadIdx.x >> 5);
  const int lane = threadIdx.x & 31;
  const float* a = x + (size_t)row * HIDm;
  float acc = a[lane] * w[lane] + a[lane + 32] * w[lane + 32] +
              a[lane + 64] * w[lane + 64] + a[lane + 96] * w[lane + 96];
  acc = wave_sum(acc);
  if (lane == 0) out[row] = acc + b[0];
}

// softmax over T=512 per batch row
__global__ __launch_bounds__(256) void softmaxT(
    const float* __restrict__ s, float* __restrict__ w) {
  const int b = blockIdx.x, tid = threadIdx.x;
  float v0 = s[b * Tt + tid], v1 = s[b * Tt + 256 + tid];
  float mx = fmaxf(v0, v1);
#pragma unroll
  for (int m = 16; m > 0; m >>= 1) mx = fmaxf(mx, __shfl_xor(mx, m, 32));
  __shared__ float sm[8];
  __shared__ float gmx_s, gsum_s;
  if ((tid & 31) == 0) sm[tid >> 5] = mx;
  __syncthreads();
  if (tid == 0) {
    float g = sm[0];
#pragma unroll
    for (int i = 1; i < 8; ++i) g = fmaxf(g, sm[i]);
    gmx_s = g;
  }
  __syncthreads();
  const float gmx = gmx_s;
  const float e0 = __expf(v0 - gmx), e1 = __expf(v1 - gmx);
  float sum = wave_sum(e0 + e1);
  if ((tid & 31) == 0) sm[tid >> 5] = sum;
  __syncthreads();
  if (tid == 0) {
    float g = 0;
#pragma unroll
    for (int i = 0; i < 8; ++i) g += sm[i];
    gsum_s = g;
  }
  __syncthreads();
  const float inv = 1.0f / gsum_s;
  w[b * Tt + tid] = e0 * inv;
  w[b * Tt + 256 + tid] = e1 * inv;
}

// attentive statistics: concat[b] = [weighted mean | weighted std]
__global__ __launch_bounds__(256) void pool_stats(
    const float* __restrict__ z, const float* __restrict__ w,
    float* __restrict__ cc) {
  const int b = blockIdx.x, tid = threadIdx.x;
  const int d0 = tid, d1 = tid + 256;
  const float* zb = z + (size_t)b * Tt * Dm;
  const float* wb = w + b * Tt;
  float m0 = 0, m1 = 0;
  for (int t = 0; t < Tt; ++t) {
    const float wt = wb[t];
    const float* zr = zb + (size_t)t * Dm;
    m0 += wt * zr[d0]; m1 += wt * zr[d1];
  }
  float v0 = 0, v1 = 0;
  for (int t = 0; t < Tt; ++t) {
    const float wt = wb[t];
    const float* zr = zb + (size_t)t * Dm;
    const float a = zr[d0] - m0, c = zr[d1] - m1;
    v0 += wt * a * a; v1 += wt * c * c;
  }
  cc[b * 2 * Dm + d0] = m0;
  cc[b * 2 * Dm + d1] = m1;
  cc[b * 2 * Dm + Dm + d0] = sqrtf(v0 + 1e-6f);
  cc[b * 2 * Dm + Dm + d1] = sqrtf(v1 + 1e-6f);
}

// small GEMV: out[m,n] = act(dot(A[m,:K], W[n,:K]) + bias[n]); wave/output
__global__ __launch_bounds__(256) void gemv_nt(
    const float* __restrict__ A, const float* __restrict__ W,
    const float* __restrict__ bias, float* __restrict__ out,
    int M, int N, int K, int act) {
  const int gid = blockIdx.x * 8 + (threadIdx.x >> 5);
  if (gid >= M * N) return;
  const int m = gid / N, n = gid % N;
  const int lane = threadIdx.x & 31;
  const float* a = A + (size_t)m * K;
  const float* w = W + (size_t)n * K;
  float acc = 0;
  for (int k = lane; k < K; k += 32) acc += a[k] * w[k];
  acc = wave_sum(acc);
  if (lane == 0) {
    float v = acc + (bias ? bias[n] : 0.0f);
    if (act == 1) v = fmaxf(v, 0.0f);
    else if (act == 2) v = tanhf(v);
    out[(size_t)m * N + n] = v;
  }
}

// one residual-VQ stage; block b handles latent row b
__global__ __launch_bounds__(256) void rvq_stage(
    const float* __restrict__ cb, float* __restrict__ resid,
    float* __restrict__ qout, float* __restrict__ codes,
    float* __restrict__ loss, int stage) {
  const int b = blockIdx.x, tid = threadIdx.x;
  __shared__ float sb[256];
  __shared__ int   si[256];
  float best = 3.0e38f; int bestj = 0;
  for (int j = tid; j < Kcb; j += 256) {
    const float* e = cb + (size_t)j * Dm;
    const float* r = resid + (size_t)b * Dm;
    float dot = 0, nrm = 0;
    for (int d = 0; d < Dm; ++d) { const float ev = e[d]; dot += r[d] * ev; nrm += ev * ev; }
    const float dist = nrm - 2.0f * dot;
    if (dist < best || (dist == best && j < bestj)) { best = dist; bestj = j; }
  }
  sb[tid] = best; si[tid] = bestj;
  __syncthreads();
  for (int s = 128; s > 0; s >>= 1) {
    if (tid < s) {
      if (sb[tid + s] < sb[tid] ||
          (sb[tid + s] == sb[tid] && si[tid + s] < si[tid])) {
        sb[tid] = sb[tid + s]; si[tid] = si[tid + s];
      }
    }
    __syncthreads();
  }
  const int j = si[0];
  __syncthreads();
  const float* e = cb + (size_t)j * Dm;
  float part = 0;
  for (int d = tid; d < Dm; d += 256) {
    const float r = resid[(size_t)b * Dm + d];
    const float ev = e[d];
    const float diff = ev - r;
    part += diff * diff;
    resid[(size_t)b * Dm + d] = r - ev;
    qout[(size_t)b * Dm + d] += ev;
  }
  sb[tid] = part;
  __syncthreads();
  for (int s = 128; s > 0; s >>= 1) {
    if (tid < s) sb[tid] += sb[tid + s];
    __syncthreads();
  }
  if (tid == 0) {
    atomicAdd(loss, COMMITF * sb[0] / (float)(Bb * Dm));
    codes[b * NCb + stage] = (float)j;
  }
}

__global__ void vzero(float* __restrict__ p, int n) {
  for (int i = blockIdx.x * blockDim.x + threadIdx.x; i < n;
       i += gridDim.x * blockDim.x) p[i] = 0.0f;
}
__global__ void vcopy(const float* __restrict__ a, float* __restrict__ o, int n) {
  for (int i = blockIdx.x * blockDim.x + threadIdx.x; i < n;
       i += gridDim.x * blockDim.x) o[i] = a[i];
}
__global__ void cvt_bf16(const float* __restrict__ s, bf16_t* __restrict__ d, int n) {
  for (int i = blockIdx.x * blockDim.x + threadIdx.x; i < n;
       i += gridDim.x * blockDim.x) d[i] = (bf16_t)s[i];
}
// t[b,t,:] = pos[t,:]
__global__ void bcast_pos(const float* __restrict__ pos, float* __restrict__ t, int n) {
  for (int i = blockIdx.x * blockDim.x + threadIdx.x; i < n;
       i += gridDim.x * blockDim.x) t[i] = pos[i % (Tt * Dm)];
}
// t[b,t,:] += v[b,:]
__global__ void add_bcast(float* __restrict__ t, const float* __restrict__ v, int n) {
  for (int i = blockIdx.x * blockDim.x + threadIdx.x; i < n;
       i += gridDim.x * blockDim.x) {
    const int b = i / (Tt * Dm);
    const int d = i % Dm;
    t[i] += v[b * Dm + d];
  }
}

__global__ __launch_bounds__(256) void recon_loss(
    const float* __restrict__ r, const float* __restrict__ x,
    float* __restrict__ loss, int n) {
  float acc = 0;
  for (int i = blockIdx.x * blockDim.x + threadIdx.x; i < n;
       i += gridDim.x * blockDim.x) {
    const float d = r[i] - x[i];
    acc += d * d;
  }
  acc = wave_sum(acc);
  __shared__ float sw[8];
  if ((threadIdx.x & 31) == 0) sw[threadIdx.x >> 5] = acc;
  __syncthreads();
  if (threadIdx.x == 0) {
    float s = 0;
#pragma unroll
    for (int i = 0; i < 8; ++i) s += sw[i];
    atomicAdd(loss, s / (float)n);
  }
}

// ---------------------------------------------------------------------------
extern "C" void kernel_launch(void* const* d_in, const int* in_sizes, int n_in,
                              void* d_out, int out_size, void* d_ws, size_t ws_size,
                              hipStream_t stream) {
  (void)in_sizes; (void)n_in; (void)out_size; (void)ws_size;
  const float* x         = (const float*)d_in[0];
  const float* in_proj_w = (const float*)d_in[1];
  const float* in_proj_b = (const float*)d_in[2];
  const float* in_norm_g = (const float*)d_in[3];
  const float* in_norm_b = (const float*)d_in[4];
  const float* feat_g    = (const float*)d_in[5];
  const float* feat_b    = (const float*)d_in[6];
  const float* enc_qkv_w = (const float*)d_in[7];
  const float* enc_qkv_b = (const float*)d_in[8];
  const float* enc_out_w = (const float*)d_in[9];
  const float* enc_out_b = (const float*)d_in[10];
  const float* enc_ln1_g = (const float*)d_in[11];
  const float* enc_ln1_b = (const float*)d_in[12];
  const float* enc_ln2_g = (const float*)d_in[13];
  const float* enc_ln2_b = (const float*)d_in[14];
  const float* enc_ff1_w = (const float*)d_in[15];
  const float* enc_ff1_b = (const float*)d_in[16];
  const float* enc_ff2_w = (const float*)d_in[17];
  const float* enc_ff2_b = (const float*)d_in[18];
  const float* att1_w    = (const float*)d_in[19];
  const float* att1_b    = (const float*)d_in[20];
  const float* att2_w    = (const float*)d_in[21];
  const float* att2_b    = (const float*)d_in[22];
  const float* pool_w    = (const float*)d_in[23];
  const float* pool_b    = (const float*)d_in[24];
  const float* codebooks = (const float*)d_in[25];
  const float* pos_q     = (const float*)d_in[26];
  const float* dsa_qkv_w = (const float*)d_in[27];
  const float* dsa_qkv_b = (const float*)d_in[28];
  const float* dsa_out_w = (const float*)d_in[29];
  const float* dsa_out_b = (const float*)d_in[30];
  const float* dca_qkv_w = (const float*)d_in[31];
  const float* dca_qkv_b = (const float*)d_in[32];
  const float* dca_out_w = (const float*)d_in[33];
  const float* dca_out_b = (const float*)d_in[34];
  const float* dec_ln1_g = (const float*)d_in[35];
  const float* dec_ln1_b = (const float*)d_in[36];
  const float* dec_ln2_g = (const float*)d_in[37];
  const float* dec_ln2_b = (const float*)d_in[38];
  const float* dec_ln3_g = (const float*)d_in[39];
  const float* dec_ln3_b = (const float*)d_in[40];
  const float* dec_ff1_w = (const float*)d_in[41];
  const float* dec_ff1_b = (const float*)d_in[42];
  const float* dec_ff2_w = (const float*)d_in[43];
  const float* dec_ff2_b = (const float*)d_in[44];
  const float* head_w    = (const float*)d_in[45];
  const float* head_b    = (const float*)d_in[46];

  float* out   = (float*)d_out;
  float* loss  = out + (size_t)NTOK * DIm;   // 12582912
  float* codes = loss + 1;

  // workspace layout (floats)
  const size_t TOKD = (size_t)NTOK * Dm;     // 8388608
  float* ws   = (float*)d_ws;
  float* hbuf = ws;
  float* qb   = ws + TOKD;
  float* kb   = ws + 2 * TOKD;
  float* vb   = ws + 3 * TOKD;
  float* bufA = ws + 4 * TOKD;
  float* bufB = ws + 5 * TOKD;
  float* big  = ws + 6 * TOKD;               // NTOK x 2048
  float* sm   = big + (size_t)NTOK * DFm;
  float* s_w   = sm;
  float* w_w   = sm + NTOK;
  float* cc_w  = sm + 2 * NTOK;
  float* zp_w  = cc_w + (size_t)Bb * 2 * Dm;
  float* rs_w  = zp_w + (size_t)Bb * Dm;
  float* qo_w  = rs_w + (size_t)Bb * Dm;
  float* cv_w  = qo_w + (size_t)Bb * Dm;
  float* cav_w = cv_w + (size_t)Bb * Dm;

  // bf16 weight arena (element offsets)
  bf16_t* wbf = (bf16_t*)(cav_w + (size_t)Bb * Dm);
  const size_t O_INPROJ = 0;                        // 512*768
  const size_t O_EQKV   = O_INPROJ + 393216;        // 4*3*512*512
  const size_t O_EOUT   = O_EQKV   + 3145728;       // 4*512*512
  const size_t O_EFF1   = O_EOUT   + 1048576;       // 4*2048*512
  const size_t O_EFF2   = O_EFF1   + 4194304;       // 4*512*2048
  const size_t O_ATT1   = O_EFF2   + 4194304;       // 128*512
  const size_t O_DQKV   = O_ATT1   + 65536;
  const size_t O_DOUT   = O_DQKV   + 3145728;
  const size_t O_DFF1   = O_DOUT   + 1048576;
  const size_t O_DFF2   = O_DFF1   + 4194304;
  const size_t O_HEAD   = O_DFF2   + 4194304;       // 768*512

  const dim3 blk(256);
  const dim3 gD(NTOK / GBM, Dm / GBN);
  const dim3 gF1(NTOK / GBM, DFm / GBN);
  const dim3 gH(NTOK / GBM, DIm / GBN);
  const dim3 gA1(NTOK / GBM, HIDm / GBN);
  const dim3 gAtt(Tt / FQT, Hh, Bb);

  vzero<<<1, 32, 0, stream>>>(loss, 1);

  // one-time (per call) weight conversion to bf16 arena
  cvt_bf16<<<1024, blk, 0, stream>>>(in_proj_w, wbf + O_INPROJ, 393216);
  cvt_bf16<<<4096, blk, 0, stream>>>(enc_qkv_w, wbf + O_EQKV, 3145728);
  cvt_bf16<<<2048, blk, 0, stream>>>(enc_out_w, wbf + O_EOUT, 1048576);
  cvt_bf16<<<4096, blk, 0, stream>>>(enc_ff1_w, wbf + O_EFF1, 4194304);
  cvt_bf16<<<4096, blk, 0, stream>>>(enc_ff2_w, wbf + O_EFF2, 4194304);
  cvt_bf16<<<256,  blk, 0, stream>>>(att1_w,    wbf + O_ATT1, 65536);
  cvt_bf16<<<4096, blk, 0, stream>>>(dsa_qkv_w, wbf + O_DQKV, 3145728);
  cvt_bf16<<<2048, blk, 0, stream>>>(dsa_out_w, wbf + O_DOUT, 1048576);
  cvt_bf16<<<4096, blk, 0, stream>>>(dec_ff1_w, wbf + O_DFF1, 4194304);
  cvt_bf16<<<4096, blk, 0, stream>>>(dec_ff2_w, wbf + O_DFF2, 4194304);
  cvt_bf16<<<1024, blk, 0, stream>>>(head_w,    wbf + O_HEAD, 393216);

  // input projection + LN
  gemm_nt<<<gD, blk, 0, stream>>>(x, wbf + O_INPROJ, in_proj_b, bufA,
                                  NTOK, Dm, DIm, 0);
  ln512<<<NTOK, blk, 0, stream>>>(bufA, nullptr, in_norm_g, in_norm_b, hbuf);

  // encoder
  for (int l = 0; l < 4; ++l) {
    const bf16_t* wqkv = wbf + O_EQKV + (size_t)l * 3 * Dm * Dm;
    const float*  bqkv = enc_qkv_b + (size_t)l * 3 * Dm;
    gemm_nt<<<gD, blk, 0, stream>>>(hbuf, wqkv,             bqkv,        qb, NTOK, Dm, Dm, 0);
    gemm_nt<<<gD, blk, 0, stream>>>(hbuf, wqkv + Dm * Dm,   bqkv + Dm,   kb, NTOK, Dm, Dm, 0);
    gemm_nt<<<gD, blk, 0, stream>>>(hbuf, wqkv + 2*Dm*Dm,   bqkv + 2*Dm, vb, NTOK, Dm, Dm, 0);
    flash_attn<<<gAtt, blk, 0, stream>>>(qb, kb, vb, bufA);
    gemm_nt<<<gD, blk, 0, stream>>>(bufA, wbf + O_EOUT + (size_t)l * Dm * Dm,
                                    enc_out_b + (size_t)l * Dm, bufB, NTOK, Dm, Dm, 0);
    ln512<<<NTOK, blk, 0, stream>>>(bufB, hbuf, enc_ln1_g + l * Dm, enc_ln1_b + l * Dm, hbuf);
    gemm_nt<<<gF1, blk, 0, stream>>>(hbuf, wbf + O_EFF1 + (size_t)l * DFm * Dm,
                                     enc_ff1_b + (size_t)l * DFm, big, NTOK, DFm, Dm, 1);
    gemm_nt<<<gD, blk, 0, stream>>>(big, wbf + O_EFF2 + (size_t)l * Dm * DFm,
                                    enc_ff2_b + (size_t)l * Dm, bufB, NTOK, Dm, DFm, 0);
    ln512<<<NTOK, blk, 0, stream>>>(bufB, hbuf, enc_ln2_g + l * Dm, enc_ln2_b + l * Dm, hbuf);
  }

  // z = LN(h) ; attentive statistics pooling
  ln512<<<NTOK, blk, 0, stream>>>(hbuf, nullptr, feat_g, feat_b, bufA);
  gemm_nt<<<gA1, blk, 0, stream>>>(bufA, wbf + O_ATT1, att1_b, qb, NTOK, HIDm, Dm, 2);
  row_dot128<<<NTOK / 8, blk, 0, stream>>>(qb, att2_w, att2_b, s_w);
  softmaxT<<<Bb, blk, 0, stream>>>(s_w, w_w);
  pool_stats<<<Bb, blk, 0, stream>>>(bufA, w_w, cc_w);
  gemv_nt<<<(Bb * Dm + 7) / 8, blk, 0, stream>>>(cc_w, pool_w, pool_b, zp_w,
                                                 Bb, Dm, 2 * Dm, 0);

  // residual VQ
  vcopy<<<64, blk, 0, stream>>>(zp_w, rs_w, Bb * Dm);
  vzero<<<64, blk, 0, stream>>>(qo_w, Bb * Dm);
  for (int i = 0; i < NCb; ++i)
    rvq_stage<<<Bb, blk, 0, stream>>>(codebooks + (size_t)i * Kcb * Dm,
                                      rs_w, qo_w, codes, loss, i);

  // decoder (t lives in hbuf)
  bcast_pos<<<2048, blk, 0, stream>>>(pos_q, hbuf, (int)TOKD);
  for (int l = 0; l < 4; ++l) {
    const bf16_t* wqkv = wbf + O_DQKV + (size_t)l * 3 * Dm * Dm;
    const float*  bqkv = dsa_qkv_b + (size_t)l * 3 * Dm;
    gemm_nt<<<gD, blk, 0, stream>>>(hbuf, wqkv,             bqkv,        qb, NTOK, Dm, Dm, 0);
    gemm_nt<<<gD, blk, 0, stream>>>(hbuf, wqkv + Dm * Dm,   bqkv + Dm,   kb, NTOK, Dm, Dm, 0);
    gemm_nt<<<gD, blk, 0, stream>>>(hbuf, wqkv + 2*Dm*Dm,   bqkv + 2*Dm, vb, NTOK, Dm, Dm, 0);
    flash_attn<<<gAtt, blk, 0, stream>>>(qb, kb, vb, bufA);
    gemm_nt<<<gD, blk, 0, stream>>>(bufA, wbf + O_DOUT + (size_t)l * Dm * Dm,
                                    dsa_out_b + (size_t)l * Dm, bufB, NTOK, Dm, Dm, 0);
    ln512<<<NTOK, blk, 0, stream>>>(bufB, hbuf, dec_ln1_g + l * Dm, dec_ln1_b + l * Dm, hbuf);

    // cross-attention over length-1 memory == broadcast of out(v(z_q))
    const float* wv_ca = dca_qkv_w + (size_t)l * 3 * Dm * Dm + (size_t)2 * Dm * Dm;
    const float* bv_ca = dca_qkv_b + (size_t)l * 3 * Dm + 2 * Dm;
    gemv_nt<<<(Bb * Dm + 7) / 8, blk, 0, stream>>>(qo_w, wv_ca, bv_ca, cv_w, Bb, Dm, Dm, 0);
    gemv_nt<<<(Bb * Dm + 7) / 8, blk, 0, stream>>>(cv_w, dca_out_w + (size_t)l * Dm * Dm,
                                                   dca_out_b + (size_t)l * Dm, cav_w,
                                                   Bb, Dm, Dm, 0);
    add_bcast<<<2048, blk, 0, stream>>>(hbuf, cav_w, (int)TOKD);
    ln512<<<NTOK, blk, 0, stream>>>(hbuf, nullptr, dec_ln2_g + l * Dm, dec_ln2_b + l * Dm, hbuf);

    gemm_nt<<<gF1, blk, 0, stream>>>(hbuf, wbf + O_DFF1 + (size_t)l * DFm * Dm,
                                     dec_ff1_b + (size_t)l * DFm, big, NTOK, DFm, Dm, 1);
    gemm_nt<<<gD, blk, 0, stream>>>(big, wbf + O_DFF2 + (size_t)l * Dm * DFm,
                                    dec_ff2_b + (size_t)l * Dm, bufB, NTOK, Dm, DFm, 0);
    ln512<<<NTOK, blk, 0, stream>>>(bufB, hbuf, dec_ln3_g + l * Dm, dec_ln3_b + l * Dm, hbuf);
  }

  // head -> x_recon directly into d_out; losses
  gemm_nt<<<gH, blk, 0, stream>>>(hbuf, wbf + O_HEAD, head_b, out, NTOK, DIm, Dm, 0);
  recon_loss<<<1024, blk, 0, stream>>>(out, x, loss, NTOK * DIm);
}
